// OrthoLinear_38671885533507
// MI455X (gfx1250) — matrix operand order repaired and checked by reading, not compiled, not run on hardware
//
#include <hip/hip_runtime.h>
#include <hip/hip_bf16.h>

// ---------------------------------------------------------------------------
// OrthoLinear for MI455X (gfx1250, wave32, WMMA)
//   out[8192,4096] = x @ W^T,  W = dequant_int4 + alpha * CSR residual
// Kernel 1: materialize dense f16 W into d_ws (row-major [out][in]).
// Kernel 2: f16 WMMA GEMM, fp32 accumulate.
// ---------------------------------------------------------------------------

typedef __attribute__((ext_vector_type(16))) _Float16 v16h;
typedef __attribute__((ext_vector_type(8)))  _Float16 v8h;
typedef __attribute__((ext_vector_type(8)))  float    v8f;

#define OUT_F 4096
#define IN_F  4096
#define M_TOK 8192

// ---------------------------------------------------------------------------
// Kernel 1: dequantize one weight row per block + scatter CSR residual.
// ---------------------------------------------------------------------------
__global__ __launch_bounds__(256) void ortho_dequant_kernel(
    const unsigned char* __restrict__ packed,   // [OUT_F][IN_F/2]
    const float*         __restrict__ scales,   // [OUT_F]
    const _Float16*      __restrict__ vals,     // [nnz]
    const int*           __restrict__ cols,     // [nnz]
    const int*           __restrict__ ptr,      // [OUT_F+1]
    const float*         __restrict__ alpha_p,  // [1]
    _Float16*            __restrict__ W)        // [OUT_F][IN_F] f16
{
    __shared__ _Float16 srow[IN_F];             // 8 KB
    const int o   = blockIdx.x;
    const int tid = threadIdx.x;
    const float scale = scales[o];

    // 8 packed bytes per thread -> 16 output elements
    const unsigned long long pk =
        *(const unsigned long long*)(packed + (size_t)o * (IN_F / 2) + (size_t)tid * 8);
#pragma unroll
    for (int i = 0; i < 8; ++i) {
        const unsigned b  = (unsigned)((pk >> (8 * i)) & 0xFFu);
        const int lo = (int)(b & 0xFu) - 8;
        const int hi = (int)(b >> 4)   - 8;
        srow[tid * 16 + 2 * i]     = (_Float16)((float)lo * scale);
        srow[tid * 16 + 2 * i + 1] = (_Float16)((float)hi * scale);
    }
    __syncthreads();

    // CSR residual for this row: positions unique -> no races
    const float alpha = alpha_p[0];
    const int beg = ptr[o], end = ptr[o + 1];
    for (int i = beg + tid; i < end; i += 256) {
        const int c = cols[i];
        srow[c] += (_Float16)(alpha * (float)vals[i]);
    }
    __syncthreads();

    // stream row to global (512 x 16B)
    v8h*       dst = (v8h*)(W + (size_t)o * IN_F);
    const v8h* src = (const v8h*)srow;
    for (int i = tid; i < IN_F / 8; i += 256) dst[i] = src[i];
}

// ---------------------------------------------------------------------------
// Kernel 2: tiled f16 WMMA GEMM.  C[M,N] = X[M,K] * W[N,K]^T
// Block tile 128x128, BK=64, 8 waves (4 in M x 2 in N), wave tile 32x64.
// ---------------------------------------------------------------------------
#define BM 128
#define BN 128
#define BK 64
#define LDA (BK + 8)   // 72 halves (144 B) -> bank-spread stride
#define LDB (BK + 8)

union FragU { v16h v; v8h h[2]; };

__global__ __launch_bounds__(256) void ortho_gemm_kernel(
    const float*    __restrict__ X,   // [M_TOK][IN_F] fp32
    const _Float16* __restrict__ W,   // [OUT_F][IN_F] f16
    float*          __restrict__ Out) // [M_TOK][OUT_F] fp32
{
    __shared__ _Float16 As[BM * LDA]; // 18 KB
    __shared__ _Float16 Bs[BN * LDB]; // 18 KB

    const int tid  = threadIdx.x;
    const int lane = tid & 31;
    const int wave = tid >> 5;
    const int wr   = wave & 3;   // wave row   (M): 4 waves * 32 rows
    const int wc   = wave >> 2;  // wave col   (N): 2 waves * 64 cols
    const int bM   = blockIdx.y * BM;
    const int bN   = blockIdx.x * BN;

    v8f acc[2][4];
#pragma unroll
    for (int mi = 0; mi < 2; ++mi)
#pragma unroll
        for (int ni = 0; ni < 4; ++ni)
            acc[mi][ni] = (v8f)0.0f;

    // global->LDS loader assignment: 2 threads per tile row
    const int arow  = tid >> 1;
    const int acol0 = (tid & 1) * 32;  // floats
    const int brow  = tid >> 1;
    const int bcol0 = (tid & 1) * 32;  // halves

    // fragment sub-lane selectors (match ISA 16-bit A / B VGPR layouts)
    const int lrow = lane & 15;
    const int asel = (lane >> 4) * 8;   // A: upper half-wave holds K+8 block
    const int bsel = (lane >> 4) * 8;   // B: upper half-wave holds K+8 block

    for (int k0 = 0; k0 < IN_F; k0 += BK) {
        // ---- stage A: fp32 -> f16 convert into LDS ----
        const float* xp = X + (size_t)(bM + arow) * IN_F + k0 + acol0;
        _Float16*    as = As + arow * LDA + acol0;
#pragma unroll
        for (int i = 0; i < 32; i += 8) {
            const float4 f0 = *(const float4*)(xp + i);
            const float4 f1 = *(const float4*)(xp + i + 4);
            v8h h;
            h[0] = (_Float16)f0.x; h[1] = (_Float16)f0.y;
            h[2] = (_Float16)f0.z; h[3] = (_Float16)f0.w;
            h[4] = (_Float16)f1.x; h[5] = (_Float16)f1.y;
            h[6] = (_Float16)f1.z; h[7] = (_Float16)f1.w;
            *(v8h*)(as + i) = h;
        }
        // ---- stage B: f16 copy into LDS ----
        const _Float16* wp = W + (size_t)(bN + brow) * IN_F + k0 + bcol0;
        _Float16*       bs = Bs + brow * LDB + bcol0;
        *(v8h*)(bs)      = *(const v8h*)(wp);
        *(v8h*)(bs + 8)  = *(const v8h*)(wp + 8);
        *(v8h*)(bs + 16) = *(const v8h*)(wp + 16);
        *(v8h*)(bs + 24) = *(const v8h*)(wp + 24);

        if (k0 + BK < IN_F) {                 // hint next K-slab into cache
            __builtin_prefetch(xp + BK, 0, 1);
            __builtin_prefetch(wp + BK, 0, 1);
        }
        __syncthreads();

        // ---- compute: 2 K-steps of 32, 8 WMMA each ----
#pragma unroll
        for (int kk = 0; kk < BK; kk += 32) {
            FragU a[2];
#pragma unroll
            for (int mi = 0; mi < 2; ++mi) {
                const _Float16* ap =
                    As + (wr * 32 + mi * 16 + lrow) * LDA + kk + asel;
                a[mi].h[0] = *(const v8h*)ap;          // K 0..7   (or 8..15)
                a[mi].h[1] = *(const v8h*)(ap + 16);   // K 16..23 (or 24..31)
            }
            FragU b[4];
#pragma unroll
            for (int ni = 0; ni < 4; ++ni) {
                const _Float16* bp =
                    Bs + (wc * 64 + ni * 16 + lrow) * LDB + kk + bsel;
                b[ni].h[0] = *(const v8h*)bp;          // K 0..7   (or 8..15)
                b[ni].h[1] = *(const v8h*)(bp + 16);   // K 16..23 (or 24..31)
            }
#pragma unroll
            for (int mi = 0; mi < 2; ++mi)
#pragma unroll
                for (int ni = 0; ni < 4; ++ni)
                    acc[mi][ni] = __builtin_amdgcn_wmma_f32_16x16x32_f16(
                        false, a[mi].v, false, b[ni].v,
                        (short)0, acc[mi][ni], false, false);
        }
        __syncthreads();
    }

    // ---- epilogue: C/D layout -> lane<16: M=j, lane>=16: M=j+8 ----
    const int orow0 = bM + wr * 32 + (lane >> 4) * 8;
    const int ocol  = bN + wc * 64 + (lane & 15);
#pragma unroll
    for (int mi = 0; mi < 2; ++mi)
#pragma unroll
        for (int ni = 0; ni < 4; ++ni) {
            float* op = Out + (size_t)(orow0 + mi * 16) * OUT_F + ocol + ni * 16;
#pragma unroll
            for (int j = 0; j < 8; ++j)
                op[(size_t)j * OUT_F] = acc[mi][ni][j];
        }
}

// ---------------------------------------------------------------------------
extern "C" void kernel_launch(void* const* d_in, const int* in_sizes, int n_in,
                              void* d_out, int out_size, void* d_ws, size_t ws_size,
                              hipStream_t stream) {
    const float*         x      = (const float*)d_in[0];
    const unsigned char* packed = (const unsigned char*)d_in[1];
    const float*         scales = (const float*)d_in[2];
    const _Float16*      vals   = (const _Float16*)d_in[3];
    const int*           cols   = (const int*)d_in[4];
    const int*           ptr    = (const int*)d_in[5];
    const float*         alpha  = (const float*)d_in[6];
    float*               out    = (float*)d_out;
    _Float16*            Wd     = (_Float16*)d_ws;  // 32 MiB dense f16 weights

    ortho_dequant_kernel<<<OUT_F, 256, 0, stream>>>(
        packed, scales, vals, cols, ptr, alpha, Wd);

    dim3 grid(OUT_F / BN, M_TOK / BM);  // (32, 64)
    ortho_gemm_kernel<<<grid, 256, 0, stream>>>(x, Wd, out);
}
